// VQ_2920577761992
// MI455X (gfx1250) — matrix operand order repaired and checked by reading, not compiled, not run on hardware
//
#include <hip/hip_runtime.h>

// ---------------------------------------------------------------------------
// VQ nearest-codebook search for MI455X (gfx1250, wave32, WMMA + async LDS).
// x: [B=32, C=64, H=64, W=64] f32 (NCHW).  codebook: [K=512, D=64] f32.
// out: [B*H*W] nearest-codebook indices (written as float per harness).
//
// argmin_k ||x - c_k||^2  ==  argmax_k ( x . c_k - 0.5*||c_k||^2 )
// GEMM N(131072) x K(512) x D(64) in bf16 WMMA w/ f32 accumulation.
// x tile staged with GLOBAL_LOAD_ASYNC_TO_LDS_B128 (d-major layout: both the
// global and LDS sides are contiguous 16B chunks -> no VGPR round-trip,
// no LDS scatter). Codebook held in LDS as bf16.
// ---------------------------------------------------------------------------

#define D_DIM 64
#define K_DIM 512
#define ROWS_PER_BLOCK 128
#define AD_STRIDE 132 // f32 per LDS d-line of the x tile ([d][n], pad vs conflicts)
#define CB_STRIDE 72  // bf16 per LDS row of the codebook (pad: conflict-free b128)

#define AS1 __attribute__((address_space(1)))
#define AS3 __attribute__((address_space(3)))

typedef __attribute__((ext_vector_type(16))) __bf16 bf16x16;
typedef __attribute__((ext_vector_type(8)))  float  f32x8;
typedef __attribute__((ext_vector_type(4)))  int    i32x4;

union Frag {
    bf16x16        v;      // WMMA operand (8 VGPRs)
    uint4          q[2];
    unsigned int   u32[8];
    unsigned short u16[16];
};

static __device__ __forceinline__ unsigned short f32_to_bf16(float f) {
    unsigned int u = __float_as_uint(f);
    unsigned int r = u + 0x7FFFu + ((u >> 16) & 1u);   // round-to-nearest-even
    return (unsigned short)(r >> 16);
}
static __device__ __forceinline__ unsigned int pack_bf16x2(float lo, float hi) {
    return (unsigned int)f32_to_bf16(lo) | ((unsigned int)f32_to_bf16(hi) << 16);
}

__global__ __launch_bounds__(256, 1)
void vq_argmin_kernel(const float* __restrict__ x,
                      const float* __restrict__ cb,
                      float* __restrict__ out,
                      int hw, int chw)
{
    // x tile, d-major: sA[d * AD_STRIDE + n], d = 0..63, n = 0..127. 33.8 KB
    __shared__ __align__(16) float          sA[D_DIM * AD_STRIDE];
    __shared__ __align__(16) unsigned short sCB[K_DIM * CB_STRIDE];         // 72.0 KB
    __shared__ __align__(16) float          sSq[K_DIM];                     //  2.0 KB

    const int tid = threadIdx.x;
    const int blk = blockIdx.x;

    // ---- Stage x tile via async global->LDS (no VGPR round-trip) ----------
    // NCHW global is d-major/n-contiguous; LDS [d][n] keeps both sides as
    // contiguous 16B chunks: 8 x GLOBAL_LOAD_ASYNC_TO_LDS_B128 per thread.
    const int n0  = blk * ROWS_PER_BLOCK;
    const int b   = n0 / hw;
    const int pos = n0 % hw;
    const float* xb = x + (size_t)b * chw + pos;
    AS3 float* sA_l = (AS3 float*)sA;
#pragma unroll
    for (int it = 0; it < 8; ++it) {
        int q  = it * 256 + tid;        // 0..2047 float4 chunks
        int d  = q >> 5;                // 64 d-lines, 32 float4 each
        int nl = (q & 31) << 2;
        __builtin_amdgcn_global_load_async_to_lds_b128(
            (AS1 i32x4*)(xb + (size_t)d * hw + nl),
            (AS3 i32x4*)(sA_l + d * AD_STRIDE + nl),
            0, 0);
    }

    // ---- Stage codebook f32 -> bf16 in LDS, plus 0.5*||c||^2 --------------
    for (int k = tid; k < K_DIM; k += 256) {
        const float4* src = (const float4*)(cb + (size_t)k * D_DIM);
        unsigned int* dst = (unsigned int*)(sCB + k * CB_STRIDE);
        float sq = 0.0f;
#pragma unroll
        for (int i = 0; i < 16; ++i) {
            float4 v = src[i];
            sq += v.x * v.x + v.y * v.y + v.z * v.z + v.w * v.w;
            dst[2 * i + 0] = pack_bf16x2(v.x, v.y);
            dst[2 * i + 1] = pack_bf16x2(v.z, v.w);
        }
        sSq[k] = 0.5f * sq;
    }

    // Barrier does not order ASYNCcnt: drain async engine first.
    asm volatile("s_wait_asynccnt 0x0" ::: "memory");
    __syncthreads();

    // ---- Per-wave GEMM + argmax ------------------------------------------
    const int lane = tid & 31;
    const int wave = tid >> 5;
    const int half = lane >> 4;   // 0: lanes 0-15, 1: lanes 16-31
    const int l16  = lane & 15;

    // A fragments (16x32 bf16, per lane K = base..base+15 contiguous,
    // base = c*32 + half*16), row n = wave*16 + l16. From d-major LDS: scalar
    // ds_load_b32 gathers, conflict-free (lanes vary n at fixed d; half-wave
    // groups hit disjoint bank ranges). One-time cost, held in registers.
    Frag afr[2];
    {
        const int row = wave * 16 + l16;
#pragma unroll
        for (int c = 0; c < 2; ++c) {
            const float* p0 = sA + (c * 32 + half * 16) * AD_STRIDE + row;
            const float* p1 = sA + (c * 32 + half * 16 + 8) * AD_STRIDE + row;
#pragma unroll
            for (int j = 0; j < 4; ++j) {
                afr[c].u32[j]     = pack_bf16x2(p0[(2 * j) * AD_STRIDE],
                                                p0[(2 * j + 1) * AD_STRIDE]);
                afr[c].u32[4 + j] = pack_bf16x2(p1[(2 * j) * AD_STRIDE],
                                                p1[(2 * j + 1) * AD_STRIDE]);
            }
        }
    }

    float bestv[8];
    int   besti[8];
#pragma unroll
    for (int r = 0; r < 8; ++r) { bestv[r] = -3.4e38f; besti[r] = 0; }

    // Per-lane byte offset within a 16-row codebook group.
    const char* cbase = (const char*)sCB + (size_t)l16 * (CB_STRIDE * 2);

    // 32 column tiles of 16 codebook entries each. Unroll x2 so the next
    // tile's ds loads + WMMAs issue under this tile's argmax VALU.
#pragma unroll 2
    for (int t = 0; t < 32; ++t) {
        const int n = (t << 4) + l16;                     // this lane's column
        const char* crow = cbase + (size_t)t * (16 * CB_STRIDE * 2);

        // B fragments (32x16 bf16): per lane, K = c*32 + half*16 .. +15 contiguous.
        Frag bfr0, bfr1;
        {
            const char* p0 = crow + half * 32;            // c = 0
            bfr0.q[0] = *(const uint4*)(p0 + 0);
            bfr0.q[1] = *(const uint4*)(p0 + 16);
            const char* p1 = crow + 64 + half * 32;       // c = 1
            bfr1.q[0] = *(const uint4*)(p1 + 0);
            bfr1.q[1] = *(const uint4*)(p1 + 16);
        }

        f32x8 acc = {0.f, 0.f, 0.f, 0.f, 0.f, 0.f, 0.f, 0.f};
        acc = __builtin_amdgcn_wmma_f32_16x16x32_bf16(
                  false, afr[0].v, false, bfr0.v, (short)0, acc, false, false);
        acc = __builtin_amdgcn_wmma_f32_16x16x32_bf16(
                  false, afr[1].v, false, bfr1.v, (short)0, acc, false, false);

        const float bias = sSq[n];                        // 0.5*||c_n||^2
#pragma unroll
        for (int r = 0; r < 8; ++r) {
            float s = acc[r] - bias;                      // row M = r + 8*half
            bool  gt = s > bestv[r];
            bestv[r] = gt ? s : bestv[r];
            besti[r] = gt ? n : besti[r];
        }
    }

    // Cross-lane argmax within each 16-lane half (C-matrix N dimension).
#pragma unroll
    for (int m = 1; m < 16; m <<= 1) {
#pragma unroll
        for (int r = 0; r < 8; ++r) {
            float ov = __shfl_xor(bestv[r], m, 32);
            int   oi = __shfl_xor(besti[r], m, 32);
            bool  gt = ov > bestv[r];
            bestv[r] = gt ? ov : bestv[r];
            besti[r] = gt ? oi : besti[r];
        }
    }

    // Lane 0 of each half holds rows M = half*8 .. half*8+7 of this wave's strip.
    if (l16 == 0) {
        float* o = out + (size_t)n0 + wave * 16 + half * 8;
#pragma unroll
        for (int r = 0; r < 8; ++r) o[r] = (float)besti[r];
    }
}

extern "C" void kernel_launch(void* const* d_in, const int* in_sizes, int n_in,
                              void* d_out, int out_size, void* d_ws, size_t ws_size,
                              hipStream_t stream) {
    (void)n_in; (void)d_ws; (void)ws_size; (void)out_size;
    const float* x  = (const float*)d_in[0];   // [32,64,64,64] f32
    const float* cb = (const float*)d_in[1];   // [512,64] f32
    float* out = (float*)d_out;                // [131072] indices as float

    const int hw   = 64 * 64;                  // H*W
    const int chw  = 64 * hw;                  // C*H*W
    const int rows = in_sizes[0] / D_DIM;      // 131072
    const int blocks = rows / ROWS_PER_BLOCK;  // 1024

    vq_argmin_kernel<<<blocks, 256, 0, stream>>>(x, cb, out, hw, chw);
}
